// MuSc_74431783240154
// MI455X (gfx1250) — matrix operand order repaired and checked by reading, not compiled, not run on hardware
//
#include <hip/hip_runtime.h>
#include <hip/hip_bf16.h>

typedef __attribute__((ext_vector_type(16))) _Float16 v16h;
typedef __attribute__((ext_vector_type(8)))  _Float16 v8h;
typedef __attribute__((ext_vector_type(8)))  float    v8f;

#define LL 2
#define BB 8
#define PH_ 28
#define PW_ 28
#define PP 784          // PH*PW
#define DD 1024
#define HH 392
#define NQT 49          // P/16 tiles
#define AROW 1032       // LDS row stride in halves: 1024 + 8 (16B pad, bank-spread)

// ---------------------------------------------------------------------------
// Kernel 1: LayerNorm over D, write f16, and sum-of-squares per patch (rset 0)
// one block (256 threads) per (l,b,p) row; 4 elements per thread
// ---------------------------------------------------------------------------
__global__ void __launch_bounds__(256) musc_ln_kernel(
    const float* __restrict__ x, _Float16* __restrict__ ln,
    float* __restrict__ sq)
{
    const int row  = blockIdx.x;                 // l*B*P + b*P + p
    const int tid  = threadIdx.x;
    const int wave = tid >> 5, lane = tid & 31;
    const float* xr = x + (size_t)row * DD;

    float v[4], s = 0.f, s2 = 0.f;
    #pragma unroll
    for (int i = 0; i < 4; ++i) {
        v[i] = xr[tid + i * 256];
        s  += v[i];
        s2 += v[i] * v[i];
    }
    #pragma unroll
    for (int off = 16; off; off >>= 1) {
        s  += __shfl_xor(s,  off, 32);
        s2 += __shfl_xor(s2, off, 32);
    }
    __shared__ float sh0[8], sh1[8];
    if (lane == 0) { sh0[wave] = s; sh1[wave] = s2; }
    __syncthreads();
    float ts = 0.f, ts2 = 0.f;
    #pragma unroll
    for (int w = 0; w < 8; ++w) { ts += sh0[w]; ts2 += sh1[w]; }
    const float mean = ts * (1.f / (float)DD);
    const float var  = ts2 * (1.f / (float)DD) - mean * mean;
    const float rstd = rsqrtf(var + 1e-5f);

    _Float16* lr = ln + (size_t)row * DD;
    float sqp = 0.f;
    #pragma unroll
    for (int i = 0; i < 4; ++i) {
        _Float16 h = (_Float16)((v[i] - mean) * rstd);
        lr[tid + i * 256] = h;
        float f = (float)h;
        sqp += f * f;
    }
    __syncthreads();
    #pragma unroll
    for (int off = 16; off; off >>= 1) sqp += __shfl_xor(sqp, off, 32);
    if (lane == 0) sh0[wave] = sqp;
    __syncthreads();
    if (tid == 0) {
        float t = 0.f;
        #pragma unroll
        for (int w = 0; w < 8; ++w) t += sh0[w];
        sq[row] = t;
    }
}

// ---------------------------------------------------------------------------
// Kernel 2: 3x3 avg pool (zero-pad, /9) on the 28x28 patch grid (rset 1)
// ---------------------------------------------------------------------------
__global__ void __launch_bounds__(256) musc_pool_kernel(
    const _Float16* __restrict__ ln, _Float16* __restrict__ rf,
    float* __restrict__ sq)
{
    const int row = blockIdx.x;                  // lb*P + p
    const int p   = row % PP;
    const int lb  = row / PP;
    const int py  = p / PW_, px = p % PW_;
    const int tid = threadIdx.x;
    const int wave = tid >> 5, lane = tid & 31;

    float acc[4] = {0.f, 0.f, 0.f, 0.f};
    #pragma unroll
    for (int dy = -1; dy <= 1; ++dy)
        #pragma unroll
        for (int dx = -1; dx <= 1; ++dx) {
            int yy = py + dy, xx = px + dx;
            if (yy < 0 || yy >= PH_ || xx < 0 || xx >= PW_) continue;
            const _Float16* src = ln + ((size_t)lb * PP + yy * PW_ + xx) * DD;
            #pragma unroll
            for (int i = 0; i < 4; ++i) acc[i] += (float)src[tid + i * 256];
        }

    _Float16* dst = rf + (size_t)row * DD;
    float sqp = 0.f;
    #pragma unroll
    for (int i = 0; i < 4; ++i) {
        _Float16 h = (_Float16)(acc[i] * (1.f / 9.f));
        dst[tid + i * 256] = h;
        float f = (float)h;
        sqp += f * f;
    }
    #pragma unroll
    for (int off = 16; off; off >>= 1) sqp += __shfl_xor(sqp, off, 32);
    __shared__ float sh[8];
    if (lane == 0) sh[wave] = sqp;
    __syncthreads();
    if (tid == 0) {
        float t = 0.f;
        #pragma unroll
        for (int w = 0; w < 8; ++w) t += sh[w];
        sq[row] = t;
    }
}

// ---------------------------------------------------------------------------
// Kernel 3: WMMA Gram + fused min-over-q epilogue.
// grid = (49 p-tiles, 256 pairs: rset*128 + l*64 + b*8 + c). Block = 8 waves.
// A tile (16 rows x 1024 f16 = 32KB, contiguous in global) is staged into LDS
// once per block via gfx1250 async-to-LDS (ASYNCcnt), with a 16B row pad so
// the 16 lanes reading 16 distinct rows hit distinct banks. Each wave then
// runs its q-tiles: acc = A(16x32 f16, from LDS) x B(32x16 f16, from global)
// over K=1024; runmin = min(runmin, sq[c,q] - 2*acc). Shuffle+LDS min-tree.
// ---------------------------------------------------------------------------
__global__ void __launch_bounds__(256) musc_gram_min_kernel(
    const _Float16* __restrict__ rf0, const _Float16* __restrict__ rf1,
    const float* __restrict__ sq, float* __restrict__ dmin)
{
    const int ptile = blockIdx.x;                // 0..48
    int pair = blockIdx.y;                       // 0..255
    const int c    = pair & 7;  pair >>= 3;
    const int b    = pair & 7;  pair >>= 3;
    const int l    = pair & 1;  pair >>= 1;
    const int rset = pair;                       // 0 or 1
    if (b == c) return;                          // uniform: EXEC stays full

    const _Float16* rf = rset ? rf1 : rf0;
    const _Float16* Am = rf + (size_t)(l * BB + b) * PP * DD;
    const _Float16* Bm = rf + (size_t)(l * BB + c) * PP * DD;
    const float* sqb = sq + ((size_t)rset * LL + l) * BB * PP + (size_t)b * PP;
    const float* sqc = sq + ((size_t)rset * LL + l) * BB * PP + (size_t)c * PP;

    const int tid  = threadIdx.x;
    const int wave = tid >> 5;
    const int lane = tid & 31;
    const int n    = lane & 15;                  // row (A) / col (B/D) index
    const int half = lane >> 4;
    const int p0   = ptile * 16;

    // ---- stage A tile into LDS with gfx1250 async-to-LDS ------------------
    __shared__ __align__(16) _Float16 Atile[16 * AROW];   // ~33 KB
    {
        const char* asrc = (const char*)(Am + (size_t)p0 * DD);  // contiguous 32KB
        const unsigned ldsbase = (unsigned)(uintptr_t)(&Atile[0]);
        #pragma unroll
        for (int j = 0; j < 8; ++j) {
            const int chunk = tid + j * 256;                // 2048 x 16B chunks
            const int arow  = chunk >> 7;                   // /128 chunks per row
            const int within = (chunk & 127) * 16;          // byte offset in row
            const unsigned la = ldsbase + (unsigned)(arow * (AROW * 2) + within);
            const unsigned long long ga =
                (unsigned long long)(uintptr_t)(asrc + chunk * 16);
            asm volatile("global_load_async_to_lds_b128 %0, %1, off"
                         :: "v"(la), "v"(ga) : "memory");
        }
        asm volatile("s_wait_asynccnt 0x0" ::: "memory");
    }
    __syncthreads();

    float runmin[8];
    #pragma unroll
    for (int v = 0; v < 8; ++v) runmin[v] = __builtin_inff();

    // A fragment from LDS: lane row m = n; lanes 0-15 hold K 0..15,
    // lanes 16-31 hold K 16..31 of each 32-wide k-step.
    const _Float16* arow_l = &Atile[n * AROW + half * 16];

    for (int qt = wave; qt < NQT; qt += 8) {
        const int q0 = qt * 16;
        // B fragment: lane col n; lanes 0-15 K 0..15, lanes 16-31 K 16..31.
        const _Float16* brow = Bm + (size_t)(q0 + n) * DD + half * 16;

        v8f acc = {};
        #pragma unroll 4
        for (int kk = 0; kk < DD; kk += 32) {
            v8h alo = *(const v8h*)(arow_l + kk);
            v8h ahi = *(const v8h*)(arow_l + kk + 8);
            v16h a = __builtin_shufflevector(alo, ahi,
                0, 1, 2, 3, 4, 5, 6, 7, 8, 9, 10, 11, 12, 13, 14, 15);
            v16h bf = *(const v16h*)(brow + kk);
            acc = __builtin_amdgcn_wmma_f32_16x16x32_f16(
                false, a, false, bf, (short)0, acc, false, false);
        }
        const float sc = sqc[q0 + n];
        #pragma unroll
        for (int v = 0; v < 8; ++v)
            runmin[v] = fminf(runmin[v], sc - 2.0f * acc[v]);
    }

    // min across the 16 lanes holding columns n=0..15 (two independent halves)
    #pragma unroll
    for (int off = 1; off < 16; off <<= 1)
        #pragma unroll
        for (int v = 0; v < 8; ++v)
            runmin[v] = fminf(runmin[v], __shfl_xor(runmin[v], off, 16));

    __shared__ float wmin[8][16];                // [wave][row m]
    if (n == 0) {
        #pragma unroll
        for (int v = 0; v < 8; ++v) wmin[wave][half * 8 + v] = runmin[v];
    }
    __syncthreads();
    if (tid < 16) {
        float mn = __builtin_inff();
        #pragma unroll
        for (int w = 0; w < 8; ++w) mn = fminf(mn, wmin[w][tid]);
        float d2 = sqb[p0 + tid] + mn;
        float dv = sqrtf(fmaxf(d2, 0.0f));
        dmin[((((size_t)rset * LL + l) * BB + b) * BB + c) * PP + p0 + tid] = dv;
    }
}

// ---------------------------------------------------------------------------
// Kernel 4: per (b,p): mean of the 2 smallest dmin over c (c!=b), averaged
// over the 4 (rset,l) combos -> patch_scores[b,p]
// ---------------------------------------------------------------------------
__global__ void __launch_bounds__(256) musc_combine_kernel(
    const float* __restrict__ dmin, float* __restrict__ patch)
{
    const int idx = blockIdx.x * 256 + threadIdx.x;   // b*P + p
    if (idx >= BB * PP) return;
    const int b = idx / PP, p = idx % PP;
    float s = 0.f;
    #pragma unroll
    for (int rl = 0; rl < 4; ++rl) {
        const float* dm = dmin + ((size_t)rl * BB * BB + b * BB) * PP + p;
        float m1 = __builtin_inff(), m2 = __builtin_inff();
        #pragma unroll
        for (int cc = 0; cc < 8; ++cc) {
            if (cc == b) continue;
            float v = dm[(size_t)cc * PP];
            if (v < m1) { m2 = m1; m1 = v; }
            else if (v < m2) m2 = v;
        }
        s += 0.5f * (m1 + m2);
    }
    patch[idx] = s * 0.25f;                      // /(L * len(R_LIST))
}

// ---------------------------------------------------------------------------
// Kernel 5: bilinear 28x28 -> 392x392 (half-pixel centers, edge clamp),
// write pixel map + per-image max. One block per image.
// ---------------------------------------------------------------------------
__global__ void __launch_bounds__(256) musc_upsample_kernel(
    const float* __restrict__ patch, float* __restrict__ out_scores,
    float* __restrict__ out_pixel)
{
    const int b = blockIdx.x;
    const float* am = patch + (size_t)b * PP;
    float* op = out_pixel + (size_t)b * HH * HH;
    float lmax = -__builtin_inff();

    for (int i = threadIdx.x; i < HH * HH; i += blockDim.x) {
        int y = i / HH, x = i % HH;
        float sy = (y + 0.5f) * (1.0f / 14.0f) - 0.5f;
        float sx = (x + 0.5f) * (1.0f / 14.0f) - 0.5f;
        float fy0 = floorf(sy), fx0 = floorf(sx);
        float wy = sy - fy0, wx = sx - fx0;
        int y0 = min(max((int)fy0, 0), PH_ - 1);
        int y1 = min(max((int)fy0 + 1, 0), PH_ - 1);
        int x0 = min(max((int)fx0, 0), PW_ - 1);
        int x1 = min(max((int)fx0 + 1, 0), PW_ - 1);
        float v00 = am[y0 * PW_ + x0], v01 = am[y0 * PW_ + x1];
        float v10 = am[y1 * PW_ + x0], v11 = am[y1 * PW_ + x1];
        float v = (1.f - wy) * ((1.f - wx) * v00 + wx * v01)
                +        wy  * ((1.f - wx) * v10 + wx * v11);
        op[i] = v;
        lmax = fmaxf(lmax, v);
    }
    #pragma unroll
    for (int off = 16; off; off >>= 1) lmax = fmaxf(lmax, __shfl_xor(lmax, off, 32));
    __shared__ float wmax[8];
    if ((threadIdx.x & 31) == 0) wmax[threadIdx.x >> 5] = lmax;
    __syncthreads();
    if (threadIdx.x == 0) {
        float m = wmax[0];
        #pragma unroll
        for (int w = 1; w < 8; ++w) m = fmaxf(m, wmax[w]);
        out_scores[b] = m;
    }
}

// ---------------------------------------------------------------------------
extern "C" void kernel_launch(void* const* d_in, const int* in_sizes, int n_in,
                              void* d_out, int out_size, void* d_ws, size_t ws_size,
                              hipStream_t stream)
{
    (void)in_sizes; (void)n_in; (void)out_size; (void)ws_size;
    const float* feats = (const float*)d_in[0];
    float* out = (float*)d_out;

    const size_t LN_HALVES = (size_t)LL * BB * PP * DD;      // 12.8M f16
    const size_t LN_BYTES  = LN_HALVES * sizeof(_Float16);   // ~25.7 MB

    char* ws = (char*)d_ws;
    _Float16* ln16 = (_Float16*)ws;                          // rset 0 features
    _Float16* rf16 = (_Float16*)(ws + LN_BYTES);             // rset 1 features
    float* sq    = (float*)(ws + 2 * LN_BYTES);              // [2][L][B][P]
    float* dmin  = sq + (size_t)2 * LL * BB * PP;            // [2][L][B][B][P]
    float* patch = dmin + (size_t)2 * LL * BB * BB * PP;     // [B][P]

    const int rows = LL * BB * PP;                           // 12544
    musc_ln_kernel<<<rows, 256, 0, stream>>>(feats, ln16, sq);
    musc_pool_kernel<<<rows, 256, 0, stream>>>(ln16, rf16, sq + (size_t)LL * BB * PP);

    dim3 ggrid(NQT, 256);                                    // 49 p-tiles x 256 pairs
    musc_gram_min_kernel<<<ggrid, 256, 0, stream>>>(ln16, rf16, sq, dmin);

    musc_combine_kernel<<<(BB * PP + 255) / 256, 256, 0, stream>>>(dmin, patch);
    musc_upsample_kernel<<<BB, 256, 0, stream>>>(patch, out, out + BB);
}
